// KPConvFPN_2173253452322
// MI455X (gfx1250) — matrix operand order repaired and checked by reading, not compiled, not run on hardware
//
#include <hip/hip_runtime.h>
#include <hip/hip_bf16.h>
#include <cstddef>

// ---------------------------------------------------------------------------
// Types for CDNA5 WMMA (wave32, 16x16x32 bf16 -> f32)
// ---------------------------------------------------------------------------
typedef __attribute__((ext_vector_type(16))) __bf16        v16bf;
typedef __attribute__((ext_vector_type(8)))  float         v8f;
typedef __attribute__((ext_vector_type(4)))  unsigned int  uint4v;
typedef __attribute__((ext_vector_type(4)))  float         float4v;

union Frag {                 // all members trivially constructible
    v16bf  v;
    uint4v q[2];
};

__device__ __forceinline__ unsigned int pack2(float lo, float hi) {
    union { __bf16 h[2]; unsigned u; } x;
    x.h[0] = (__bf16)lo;                 // native f32->bf16 convert (RNE)
    x.h[1] = (__bf16)hi;
    return x.u;
}

#define LEAKY(v) ((v) >= 0.0f ? (v) : 0.1f * (v))

// ---------------------------------------------------------------------------
// GEMM: C[M,N] = A[M,K] @ B[K,N] (+ bias[N] optional), fp32 in/out, bf16 WMMA.
// Block = 256 threads = 8 waves; tile 64(M) x 64(N); K stepped by 32.
// Wave w: wm=w&3, wn=w>>2; computes two 16x16 tiles (wm*16, wn*32 + {0,16})
// sharing one A fragment -> 2 WMMAs per wave per K-step.
// LDS double-buffered (one barrier per K-step); bf16 pairs packed in dwords:
//   As[buf][row][kp] : kp-th K-pair of A row     (A fragment = 2x ds_load_b128)
//   Bs[buf][col][kp] : kp-th K-pair of B column  (B fragment = 2x ds_load_b128)
// Row stride 20 dwords (80B) keeps all fragment reads 16B aligned.
// ---------------------------------------------------------------------------
__global__ __launch_bounds__(256)
void gemm_bf16_wmma(const float* __restrict__ A, const float* __restrict__ B,
                    const float* __restrict__ bias, float* __restrict__ C,
                    int M, int N, int K)
{
    __shared__ __align__(16) unsigned int As[2][64][20];
    __shared__ __align__(16) unsigned int Bs[2][64][20];

    const int tid  = threadIdx.x;
    const int m0   = blockIdx.y * 64;
    const int n0   = blockIdx.x * 64;
    const int wave = tid >> 5;
    const int lane = tid & 31;
    const int r    = lane & 15;
    const int hi   = lane >> 4;
    const int wm   = wave & 3;
    const int wn   = wave >> 2;

    // staging maps
    const int sar = tid >> 2;             // A row 0..63
    const int sak = (tid & 3) * 4;        // A kpair base 0,4,8,12
    const int sbk = tid >> 4;             // B kpair 0..15
    const int sbc = (tid & 15) * 4;       // B col base 0..60

    // loop-invariant path selection (K multiple of 32 in all but one GEMM)
    const bool fastA = ((K & 31) == 0) && (m0 + 64 <= M);
    const bool fastB = ((K & 31) == 0) && (n0 + 64 <= N) && ((N & 3) == 0);

    auto stageA = [&](int k0, int buf) {
        if (fastA) {
            const float* base = &A[(size_t)(m0 + sar) * K + k0 + sak * 2];
            const float4v v0 = *reinterpret_cast<const float4v*>(base);
            const float4v v1 = *reinterpret_cast<const float4v*>(base + 4);
            uint4v p;
            p.x = pack2(v0.x, v0.y); p.y = pack2(v0.z, v0.w);
            p.z = pack2(v1.x, v1.y); p.w = pack2(v1.z, v1.w);
            *reinterpret_cast<uint4v*>(&As[buf][sar][sak]) = p;
        } else {
            const int gm  = m0 + sar;
            const int gmc = gm < M ? gm : (M - 1);
            #pragma unroll
            for (int j = 0; j < 4; ++j) {
                int k  = k0 + (sak + j) * 2;
                int c0 = k < K ? k : 0;
                int c1 = (k + 1) < K ? (k + 1) : 0;
                float lo = A[(size_t)gmc * K + c0];
                float h  = A[(size_t)gmc * K + c1];
                lo = (gm < M && k < K)       ? lo : 0.0f;   // clamp-and-select
                h  = (gm < M && (k + 1) < K) ? h  : 0.0f;
                As[buf][sar][sak + j] = pack2(lo, h);
            }
        }
    };
    auto stageB = [&](int k0, int buf) {
        if (fastB) {
            const float4v v0 = *reinterpret_cast<const float4v*>(
                &B[(size_t)(k0 + sbk * 2) * N + n0 + sbc]);
            const float4v v1 = *reinterpret_cast<const float4v*>(
                &B[(size_t)(k0 + sbk * 2 + 1) * N + n0 + sbc]);
            Bs[buf][sbc    ][sbk] = pack2(v0.x, v1.x);
            Bs[buf][sbc + 1][sbk] = pack2(v0.y, v1.y);
            Bs[buf][sbc + 2][sbk] = pack2(v0.z, v1.z);
            Bs[buf][sbc + 3][sbk] = pack2(v0.w, v1.w);
        } else {
            const int kA  = k0 + sbk * 2;
            const int kB  = kA + 1;
            const int kAc = kA < K ? kA : 0;
            const int kBc = kB < K ? kB : 0;
            #pragma unroll
            for (int j = 0; j < 4; ++j) {
                int gn  = n0 + sbc + j;
                int gnc = gn < N ? gn : (N - 1);
                float lo = B[(size_t)kAc * N + gnc];
                float h  = B[(size_t)kBc * N + gnc];
                lo = (kA < K && gn < N) ? lo : 0.0f;
                h  = (kB < K && gn < N) ? h  : 0.0f;
                Bs[buf][sbc + j][sbk] = pack2(lo, h);
            }
        }
    };

    v8f acc0 = {}, acc1 = {};
    const int nk = (K + 31) >> 5;

    stageA(0, 0);
    stageB(0, 0);
    __syncthreads();

    for (int i = 0; i < nk; ++i) {
        const int buf = i & 1;
        if (i + 1 < nk) {                       // prefetch next tile to other buffer
            stageA((i + 1) << 5, buf ^ 1);
            stageB((i + 1) << 5, buf ^ 1);
        }
        if (fastA && ((i + 2) << 5) < K) {      // warm caches two tiles ahead
            __builtin_prefetch(&A[(size_t)(m0 + sar) * K + ((i + 2) << 5) + sak * 2], 0, 0);
            if (fastB)
                __builtin_prefetch(&B[(size_t)(((i + 2) << 5) + sbk * 2) * N + n0 + sbc], 0, 0);
        }

        Frag af, bf0, bf1;
        const int fr = wm * 16 + r;
        const int ab = hi * 4;
        af.q[0] = *reinterpret_cast<const uint4v*>(&As[buf][fr][ab]);
        af.q[1] = *reinterpret_cast<const uint4v*>(&As[buf][fr][ab + 8]);
        const int c0 = wn * 32 + r;
        const int bb = hi * 4;
        bf0.q[0] = *reinterpret_cast<const uint4v*>(&Bs[buf][c0][bb]);
        bf0.q[1] = *reinterpret_cast<const uint4v*>(&Bs[buf][c0][bb + 8]);
        bf1.q[0] = *reinterpret_cast<const uint4v*>(&Bs[buf][c0 + 16][bb]);
        bf1.q[1] = *reinterpret_cast<const uint4v*>(&Bs[buf][c0 + 16][bb + 8]);

        acc0 = __builtin_amdgcn_wmma_f32_16x16x32_bf16(
            false, af.v, false, bf0.v, (short)0, acc0, false, false);
        acc1 = __builtin_amdgcn_wmma_f32_16x16x32_bf16(
            false, af.v, false, bf1.v, (short)0, acc1, false, false);

        __syncthreads();                        // single barrier per K-step
    }

    #pragma unroll
    for (int j = 0; j < 8; ++j) {
        int row  = m0 + wm * 16 + (hi ? 8 + j : j);
        int colA = n0 + wn * 32 + r;
        int colB = colA + 16;
        if (row < M) {
            if (colA < N) {
                float v = acc0[j];
                if (bias != nullptr) v += bias[colA];
                C[(size_t)row * N + colA] = v;
            }
            if (colB < N) {
                float v = acc1[j];
                if (bias != nullptr) v += bias[colB];
                C[(size_t)row * N + colB] = v;
            }
        }
    }
}

// ---------------------------------------------------------------------------
// KPConv gather: wf[m, p*Cin+c] = sum_k w(m,k,p) * feat(nbr[m,k], c)
// cnt[m] = max(1, #k with sum_c feat > 0). One workgroup per query point.
// Neighbor feature rows are fetched with the CDNA5 async global->LDS engine
// (GLOBAL_LOAD_ASYNC_TO_LDS_*, tracked by ASYNCcnt): the neighbor index is
// block-uniform, so valid/invalid is a uniform branch (zero-fill on shadow).
// ---------------------------------------------------------------------------
template <int CIN>
__global__ __launch_bounds__(128)
void kpconv_gather_t(const float* __restrict__ qp, const float* __restrict__ sp,
                     const float* __restrict__ sf, const int* __restrict__ nbr,
                     const float* __restrict__ kp, float inv_sigma,
                     int Ns, float* __restrict__ wf, float* __restrict__ cnt)
{
    constexpr int PC = 15 * CIN;
    __shared__ __align__(16) float wfacc[PC];
    __shared__ __align__(16) float nf[CIN];
    __shared__ float wsh[16];
    __shared__ float kps[48];
    __shared__ float qpt[4];
    __shared__ float nsum;
    __shared__ int   cntv;

    const int m   = blockIdx.x;
    const int tid = threadIdx.x;

    for (int i = tid; i < PC; i += 128) wfacc[i] = 0.0f;
    if (tid < 45) kps[tid] = kp[tid];
    if (tid < 3)  qpt[tid] = qp[(size_t)m * 3 + tid];
    if (tid == 0) cntv = 0;
    __syncthreads();

    for (int k = 0; k < 32; ++k) {
        const int  idx   = nbr[(size_t)m * 32 + k];
        const bool valid = (idx >= 0) && (idx < Ns);     // block-uniform
        const int  idxc  = valid ? idx : 0;
        if (tid == 0) nsum = 0.0f;

        if (valid) {
            if ((CIN & 3) == 0) {
                for (int c = tid * 4; c < CIN; c += 512) {
                    unsigned ldsoff = (unsigned)(size_t)(&nf[c]);
                    const float* g  = sf + (size_t)idxc * CIN + c;
                    asm volatile("global_load_async_to_lds_b128 %0, %1, off"
                                 :: "v"(ldsoff), "v"(g) : "memory");
                }
            } else {
                if (tid < CIN) {
                    unsigned ldsoff = (unsigned)(size_t)(&nf[tid]);
                    const float* g  = sf + (size_t)idxc * CIN + tid;
                    asm volatile("global_load_async_to_lds_b32 %0, %1, off"
                                 :: "v"(ldsoff), "v"(g) : "memory");
                }
            }
        } else {
            for (int c = tid; c < CIN; c += 128) nf[c] = 0.0f;
        }

        if (tid < 15) {
            float rx = sp[(size_t)idxc * 3 + 0] - qpt[0];
            float ry = sp[(size_t)idxc * 3 + 1] - qpt[1];
            float rz = sp[(size_t)idxc * 3 + 2] - qpt[2];
            float dx = rx - kps[tid * 3 + 0];
            float dy = ry - kps[tid * 3 + 1];
            float dz = rz - kps[tid * 3 + 2];
            float d  = sqrtf(dx * dx + dy * dy + dz * dz);
            float w  = fmaxf(1.0f - d * inv_sigma, 0.0f);
            wsh[tid] = valid ? w : 0.0f;
        }

        asm volatile("s_wait_asynccnt 0x0" ::: "memory");
        __syncthreads();                 // nf ready, nsum reset visible

        float psum = 0.0f;
        for (int c = tid; c < CIN; c += 128) psum += nf[c];
        if (psum != 0.0f) atomicAdd(&nsum, psum);
        __syncthreads();                 // nsum final

        if (tid == 0 && nsum > 0.0f) cntv++;
        if (CIN <= 128) {
            constexpr int TPG = 128 / (CIN < 1 ? 1 : CIN);
            const int c  = tid & (CIN - 1);
            const int p0 = tid / (CIN < 1 ? 1 : CIN);
            #pragma unroll
            for (int p = p0; p < 15; p += TPG)
                wfacc[p * CIN + c] += wsh[p] * nf[c];
        } else {
            #pragma unroll
            for (int p = 0; p < 15; ++p)
                for (int c = tid; c < CIN; c += 128)
                    wfacc[p * CIN + c] += wsh[p] * nf[c];
        }
        __syncthreads();                 // before next iteration's nf overwrite
    }

    if ((CIN & 3) == 0) {
        float4v* dst = reinterpret_cast<float4v*>(wf + (size_t)m * PC);
        const float4v* src = reinterpret_cast<const float4v*>(wfacc);
        for (int i = tid; i < PC / 4; i += 128) dst[i] = src[i];
    } else {
        for (int i = tid; i < PC; i += 128) wf[(size_t)m * PC + i] = wfacc[i];
    }
    if (tid == 0) cnt[m] = (float)(cntv > 1 ? cntv : 1);
}

// ---------------------------------------------------------------------------
// GroupNorm (pack mode, G=32): stats per group over all points. C power of 2.
// ---------------------------------------------------------------------------
__global__ __launch_bounds__(256)
void gn_stats_kernel(const float* __restrict__ X, int M, int C, int cgshift,
                     float* __restrict__ stats)
{
    __shared__ float ssum[256], ssq[256];
    const int g     = blockIdx.x;
    const int cg    = 1 << cgshift;
    const int total = M << cgshift;
    float s = 0.0f, q = 0.0f;
    for (int i = threadIdx.x; i < total; i += 256) {
        int n = i >> cgshift;
        int c = (g << cgshift) + (i & (cg - 1));
        float v = X[(size_t)n * C + c];
        s += v; q += v * v;
    }
    ssum[threadIdx.x] = s; ssq[threadIdx.x] = q;
    __syncthreads();
    for (int st = 128; st > 0; st >>= 1) {
        if (threadIdx.x < st) {
            ssum[threadIdx.x] += ssum[threadIdx.x + st];
            ssq[threadIdx.x]  += ssq[threadIdx.x + st];
        }
        __syncthreads();
    }
    if (threadIdx.x == 0) {
        float mean = ssum[0] / (float)total;
        float var  = ssq[0] / (float)total - mean * mean;
        stats[g]      = mean;
        stats[32 + g] = rsqrtf(var + 1e-5f);
    }
}

__global__ void gn_apply_kernel(float* __restrict__ X, int total, int cmask, int gshift,
                                const float* __restrict__ gamma, const float* __restrict__ beta,
                                const float* __restrict__ stats, int relu)
{
    int i = blockIdx.x * blockDim.x + threadIdx.x;
    if (i >= total) return;
    int c = i & cmask;
    int g = c >> gshift;
    float v = (X[i] - stats[g]) * stats[32 + g] * gamma[c] + beta[c];
    if (relu) v = LEAKY(v);
    X[i] = v;
}

// ---------------------------------------------------------------------------
// Elementwise / gather helpers (C is a power of two in every use below)
// ---------------------------------------------------------------------------
__global__ void scale_rows_kernel(float* __restrict__ X, const float* __restrict__ cnt,
                                  int total, int cshift)
{
    int i = blockIdx.x * blockDim.x + threadIdx.x;
    if (i < total) X[i] /= cnt[i >> cshift];
}

__global__ void add_leaky_kernel(const float* __restrict__ a, const float* __restrict__ b,
                                 float* __restrict__ out, int n)
{
    int i = blockIdx.x * blockDim.x + threadIdx.x;
    if (i < n) { float v = a[i] + b[i]; out[i] = LEAKY(v); }
}

__global__ void maxpool_kernel(const float* __restrict__ f, const int* __restrict__ nbr,
                               float* __restrict__ out, int total, int Ns, int cshift)
{
    int i = blockIdx.x * blockDim.x + threadIdx.x;
    if (i >= total) return;
    int m = i >> cshift, c = i & ((1 << cshift) - 1);
    const int* row = nbr + (size_t)m * 32;
    float best = -3.4e38f;
    #pragma unroll 4
    for (int k = 0; k < 32; ++k) {
        int s = row[k];
        bool valid = (s >= 0 && s < Ns);
        int  sc = valid ? s : 0;
        float v = f[((size_t)sc << cshift) + c];
        best = fmaxf(best, valid ? v : 0.0f);          // zero pad row
    }
    out[i] = best;
}

__global__ void gather_rows_kernel(const float* __restrict__ src, const int* __restrict__ idx,
                                   float* __restrict__ dst, int total, int Ns, int cshift,
                                   int dstride, int coff)
{
    int i = blockIdx.x * blockDim.x + threadIdx.x;
    if (i >= total) return;
    int m = i >> cshift, c = i & ((1 << cshift) - 1);
    int s = idx[m];
    bool valid = (s >= 0 && s < Ns);
    int  sc = valid ? s : 0;
    float v = src[((size_t)sc << cshift) + c];
    dst[(size_t)m * dstride + coff + c] = valid ? v : 0.0f;
}

__global__ void copy_cols_kernel(float* __restrict__ dst, int dstride, int coff,
                                 const float* __restrict__ src, int total, int cshift)
{
    int i = blockIdx.x * blockDim.x + threadIdx.x;
    if (i >= total) return;
    int m = i >> cshift, c = i & ((1 << cshift) - 1);
    dst[(size_t)m * dstride + coff + c] = src[i];
}

// ---------------------------------------------------------------------------
// Host-side orchestration
// ---------------------------------------------------------------------------
static inline int ilog2(int x) { return 31 - __builtin_clz((unsigned)x); }

static inline void launch_gemm(hipStream_t st, const float* A, const float* B,
                               const float* bias, float* C, int M, int N, int K)
{
    dim3 grid((N + 63) / 64, (M + 63) / 64);
    gemm_bf16_wmma<<<grid, 256, 0, st>>>(A, B, bias, C, M, N, K);
}

static inline void gn(hipStream_t st, float* X, int M, int C,
                      const float* gamma, const float* beta, float* stats, int relu)
{
    gn_stats_kernel<<<32, 256, 0, st>>>(X, M, C, ilog2(C) - 5, stats);
    int total = M * C;
    gn_apply_kernel<<<(total + 255) / 256, 256, 0, st>>>(X, total, C - 1, ilog2(C) - 5,
                                                         gamma, beta, stats, relu);
}

struct Scratch {
    float *midA, *midB, *u2, *scmp, *scu, *wf, *cnt, *stats;
};

static inline void unary_blk(hipStream_t st, const float* X, float* Y, int M, int Cin, int Cout,
                             const float* w, const float* b, const float* gamma,
                             const float* beta, float* stats, int relu)
{
    launch_gemm(st, X, w, b, Y, M, Cout, Cin);
    gn(st, Y, M, Cout, gamma, beta, stats, relu);
}

static inline void conv_blk(hipStream_t st, const float* qp, const float* sp, const float* sf,
                            const int* nbrs, const float* kpts, const float* w, float inv_sigma,
                            int Mq, int Ns, int Cin, int Cout, const float* gamma,
                            const float* beta, float* out, Scratch& ws)
{
    switch (Cin) {
    case 1:   kpconv_gather_t<1>  <<<Mq, 128, 0, st>>>(qp, sp, sf, nbrs, kpts, inv_sigma, Ns, ws.wf, ws.cnt); break;
    case 32:  kpconv_gather_t<32> <<<Mq, 128, 0, st>>>(qp, sp, sf, nbrs, kpts, inv_sigma, Ns, ws.wf, ws.cnt); break;
    case 64:  kpconv_gather_t<64> <<<Mq, 128, 0, st>>>(qp, sp, sf, nbrs, kpts, inv_sigma, Ns, ws.wf, ws.cnt); break;
    case 128: kpconv_gather_t<128><<<Mq, 128, 0, st>>>(qp, sp, sf, nbrs, kpts, inv_sigma, Ns, ws.wf, ws.cnt); break;
    default:  kpconv_gather_t<256><<<Mq, 128, 0, st>>>(qp, sp, sf, nbrs, kpts, inv_sigma, Ns, ws.wf, ws.cnt); break;
    }
    launch_gemm(st, ws.wf, w, nullptr, out, Mq, Cout, 15 * Cin);
    scale_rows_kernel<<<(Mq * Cout + 255) / 256, 256, 0, st>>>(out, ws.cnt, Mq * Cout, ilog2(Cout));
    gn(st, out, Mq, Cout, gamma, beta, ws.stats, 1);
}

struct ResP {  // pointers in sorted-leaf order per sub-dict
    const float *c_beta, *c_gamma, *c_kp, *c_w;          // conv (_kp)
    const float *sc_b, *sc_beta, *sc_gamma, *sc_w;       // sc (_lin) or null
    const float *u1_b, *u1_beta, *u1_gamma, *u1_w;       // u1 (_lin) or null
    const float *u2_b, *u2_beta, *u2_gamma, *u2_w;       // u2 (_lin)
};

static inline void residual(hipStream_t st, const ResP& p, const float* qp, const float* sp,
                            const float* fin, const int* idx, float inv_sigma,
                            int Mq, int Ns, int cin, int mid, int cout, bool strided,
                            float* out, Scratch& ws)
{
    const float* x = fin;
    if (p.u1_w) {
        unary_blk(st, fin, ws.midA, Ns, cin, mid, p.u1_w, p.u1_b, p.u1_gamma, p.u1_beta, ws.stats, 1);
        x = ws.midA;
    }
    conv_blk(st, qp, sp, x, idx, p.c_kp, p.c_w, inv_sigma, Mq, Ns, mid, mid,
             p.c_gamma, p.c_beta, ws.midB, ws);
    unary_blk(st, ws.midB, ws.u2, Mq, mid, cout, p.u2_w, p.u2_b, p.u2_gamma, p.u2_beta, ws.stats, 0);

    const float* sc = fin;
    if (strided) {
        maxpool_kernel<<<(Mq * cin + 255) / 256, 256, 0, st>>>(fin, idx, ws.scmp, Mq * cin, Ns, ilog2(cin));
        sc = ws.scmp;
    }
    if (p.sc_w) {
        unary_blk(st, sc, ws.scu, Mq, cin, cout, p.sc_w, p.sc_b, p.sc_gamma, p.sc_beta, ws.stats, 0);
        sc = ws.scu;
    }
    add_leaky_kernel<<<(Mq * cout + 255) / 256, 256, 0, st>>>(ws.u2, sc, out, Mq * cout);
}

extern "C" void kernel_launch(void* const* d_in, const int* in_sizes, int n_in,
                              void* d_out, int out_size, void* d_ws, size_t ws_size,
                              hipStream_t stream)
{
    (void)n_in; (void)out_size; (void)ws_size;
    const int N0 = 40960, N1 = 10240, N2 = 2560, N3 = 640;
    const float is0 = 1.0f / 0.05f, is1 = 1.0f / 0.10f, is2 = 1.0f / 0.20f, is3 = 1.0f / 0.40f;

    // ---- input pointer mapping (JAX tree_leaves sorts dict keys inside params) ----
    int pb;                             // base index of the 146 param leaves
    const float *feats, *pts[4];
    const int *nbrs[4], *subs[3], *up1p, *up2p;
    if (in_sizes[0] == 40960) {
        // fully key-sorted top level: feats, neighbors0-3, params..., points0-3, sub0-2, up1, up2
        feats = (const float*)d_in[0];
        for (int i = 0; i < 4; ++i) nbrs[i] = (const int*)d_in[1 + i];
        pb = 5;
        for (int i = 0; i < 4; ++i) pts[i] = (const float*)d_in[151 + i];
    } else {
        // insertion order top level: params..., feats, points0-3, neighbors0-3, sub0-2, up1, up2
        pb = 0;
        feats = (const float*)d_in[146];
        for (int i = 0; i < 4; ++i) pts[i]  = (const float*)d_in[147 + i];
        for (int i = 0; i < 4; ++i) nbrs[i] = (const int*)d_in[151 + i];
    }
    for (int i = 0; i < 3; ++i) subs[i] = (const int*)d_in[155 + i];
    up1p = (const int*)d_in[158];
    up2p = (const int*)d_in[159];

    auto PP = [&](int i) { return (const float*)d_in[pb + i]; };
    #define LIN4(i)  PP(i), PP((i)+1), PP((i)+2), PP((i)+3)          // b, beta, gamma, w
    #define CONV4(i) PP(i), PP((i)+1), PP((i)+2), PP((i)+3)          // beta, gamma, kp, w
    #define NUL4 nullptr, nullptr, nullptr, nullptr

    // ---- workspace carving (fp32) ----
    float* w = (float*)d_ws;
    size_t off = 0;
    auto alloc = [&](size_t n) { float* p = w + off; off += n; return p; };
    float* SKIP1 = alloc((size_t)N0 * 128);
    float* SKIP2 = alloc((size_t)N1 * 256);
    float* SKIP3 = alloc((size_t)N2 * 512);
    float* FA    = alloc((size_t)N0 * 64);
    float* FB    = alloc((size_t)N0 * 64);
    Scratch ws_;
    ws_.midA  = alloc((size_t)N0 * 32);
    ws_.midB  = alloc((size_t)N0 * 32);
    ws_.u2    = alloc((size_t)N0 * 128);
    ws_.scmp  = alloc((size_t)N0 * 128);
    ws_.scu   = alloc((size_t)N0 * 128);
    ws_.wf    = alloc((size_t)N0 * 480);      // also reused as concat buffer
    ws_.cnt   = alloc((size_t)N0);
    ws_.stats = alloc(64);

    float* OUT1 = (float*)d_out;                                 // 10240 x 256
    float* OUT2 = OUT1 + (size_t)N1 * 256;                       // 2560  x 512
    float* OUT3 = OUT2 + (size_t)N2 * 512;                       // 640   x 1024
    hipStream_t st = stream;

    // ---- encoder ----
    conv_blk(st, pts[0], pts[0], feats, nbrs[0], PP(8), PP(9), is0,
             N0, N0, 1, 64, PP(7), PP(6), FA, ws_);              // e1_1: 1 -> 64

    ResP r;
    r = (ResP){ CONV4(10), LIN4(14), LIN4(18), LIN4(22) };       // e1_2: 64->128 mid32
    residual(st, r, pts[0], pts[0], FA, nbrs[0], is0, N0, N0, 64, 32, 128, false, SKIP1, ws_);

    r = (ResP){ CONV4(26), NUL4, LIN4(30), LIN4(34) };           // e2_1: 128->128 mid32, strided
    residual(st, r, pts[1], pts[0], SKIP1, subs[0], is0, N1, N0, 128, 32, 128, true, FA, ws_);

    r = (ResP){ CONV4(38), LIN4(42), LIN4(46), LIN4(50) };       // e2_2: 128->256 mid64
    residual(st, r, pts[1], pts[1], FA, nbrs[1], is1, N1, N1, 128, 64, 256, false, FB, ws_);

    r = (ResP){ CONV4(54), NUL4, LIN4(58), LIN4(62) };           // e2_3: 256->256 mid64
    residual(st, r, pts[1], pts[1], FB, nbrs[1], is1, N1, N1, 256, 64, 256, false, SKIP2, ws_);

    r = (ResP){ CONV4(66), NUL4, LIN4(70), LIN4(74) };           // e3_1: 256->256 mid64, strided
    residual(st, r, pts[2], pts[1], SKIP2, subs[1], is1, N2, N1, 256, 64, 256, true, FA, ws_);

    r = (ResP){ CONV4(78), LIN4(82), LIN4(86), LIN4(90) };       // e3_2: 256->512 mid128
    residual(st, r, pts[2], pts[2], FA, nbrs[2], is2, N2, N2, 256, 128, 512, false, FB, ws_);

    r = (ResP){ CONV4(94), NUL4, LIN4(98), LIN4(102) };          // e3_3: 512->512 mid128
    residual(st, r, pts[2], pts[2], FB, nbrs[2], is2, N2, N2, 512, 128, 512, false, SKIP3, ws_);

    r = (ResP){ CONV4(106), NUL4, LIN4(110), LIN4(114) };        // e4_1: 512->512 mid128, strided
    residual(st, r, pts[3], pts[2], SKIP3, subs[2], is2, N3, N2, 512, 128, 512, true, FA, ws_);

    r = (ResP){ CONV4(118), LIN4(122), LIN4(126), LIN4(130) };   // e4_2: 512->1024 mid256
    residual(st, r, pts[3], pts[3], FA, nbrs[3], is3, N3, N3, 512, 256, 1024, false, FB, ws_);

    r = (ResP){ CONV4(134), NUL4, LIN4(138), LIN4(142) };        // e4_3: 1024->1024 mid256
    residual(st, r, pts[3], pts[3], FB, nbrs[3], is3, N3, N3, 1024, 256, 1024, false, OUT3, ws_);

    // ---- decoder ----
    float* CAT = ws_.wf;   // big arena: 2560x1536 and 10240x768 both fit

    gather_rows_kernel<<<(N2 * 1024 + 255) / 256, 256, 0, st>>>(OUT3, up2p, CAT, N2 * 1024, N3, 10, 1536, 0);
    copy_cols_kernel  <<<(N2 * 512  + 255) / 256, 256, 0, st>>>(CAT, 1536, 1024, SKIP3, N2 * 512, 9);
    launch_gemm(st, CAT, PP(5), PP(2), OUT2, N2, 512, 1536);
    gn(st, OUT2, N2, 512, PP(4), PP(3), ws_.stats, 1);

    gather_rows_kernel<<<(N1 * 512 + 255) / 256, 256, 0, st>>>(OUT2, up1p, CAT, N1 * 512, N2, 9, 768, 0);
    copy_cols_kernel  <<<(N1 * 256 + 255) / 256, 256, 0, st>>>(CAT, 768, 512, SKIP2, N1 * 256, 8);
    launch_gemm(st, CAT, PP(1), PP(0), OUT1, N1, 256, 768);

    #undef LIN4
    #undef CONV4
    #undef NUL4
}
